// GCN_3461743640613
// MI455X (gfx1250) — matrix operand order repaired and checked by reading, not compiled, not run on hardware
//
#include <hip/hip_runtime.h>
#include <cstdint>
#include <cstddef>

#define N_FEAT 256
#define HIDDEN 256
#define HID2   128

typedef __attribute__((ext_vector_type(16))) __bf16 bf16x16;
typedef __attribute__((ext_vector_type(8)))  float  f32x8;

struct Uint4Pair { uint4 a; uint4 b; };

__device__ __forceinline__ unsigned short f2bf(float f) {
  unsigned int u = __float_as_uint(f);
  u += 0x7FFFu + ((u >> 16) & 1u);          // round-to-nearest-even
  return (unsigned short)(u >> 16);
}

// ---------------- degree / norm ----------------
__global__ void fill_f32(float* __restrict__ p, float v, int n) {
  int i = blockIdx.x * blockDim.x + threadIdx.x;
  if (i < n) p[i] = v;
}

__global__ void accum_deg(const long long* __restrict__ dst, float* __restrict__ deg, int E) {
  int i = blockIdx.x * blockDim.x + threadIdx.x;
  if (i < E) unsafeAtomicAdd(&deg[(int)dst[i]], 1.0f);
}

__global__ void rsqrt_ip(float* __restrict__ p, int n) {
  int i = blockIdx.x * blockDim.x + threadIdx.x;
  if (i < n) p[i] = rsqrtf(p[i]);           // deg >= 1 (self loop), no zero guard needed
}

// ---------------- W pre-swizzle to B-fragment layout ----------------
// Tile (kt, nt) of 32x16; lane l holds column n = nt*16 + (l&15),
// K values kt*32 + (l<16 ? 0..7,16..23 : 8..15,24..31), stored contiguously (one 32B load).
__global__ void swizzle_w(const float* __restrict__ W, unsigned short* __restrict__ out,
                          int K, int N) {
  int tile = blockIdx.x;                    // tile = kt * (N/16) + nt
  int ntile = N >> 4;
  int nt = tile % ntile, kt = tile / ntile;
  int lane = threadIdx.x;
  int n = nt * 16 + (lane & 15);
  int kbase = kt * 32 + (lane >> 4) * 8;
  unsigned short* dst = out + (size_t)tile * 512 + lane * 16;
#pragma unroll
  for (int j = 0; j < 16; ++j) dst[j] = f2bf(W[(size_t)(kbase + j + (j & 8)) * N + n]);
}

// ---------------- WMMA GEMM: C[M,NCOLS] = A[M,256] x B, bf16 in / f32 acc ----------------
// 128 threads = 4 waves; each block does 16 rows x NCOLS; wave w covers TPW 16-col tiles.
template <int NCOLS, int TPW>
__global__ __launch_bounds__(128) void gemm_bf16_wmma(
    const float* __restrict__ A, const unsigned short* __restrict__ Bswz,
    float* __restrict__ C) {
  constexpr int K  = 256;
  constexpr int KT = K / 32;
  constexpr int NT = NCOLS / 16;
  __shared__ unsigned short As[2][16][40];  // 80B pitch: conflict-free b128 reads

  const int tid  = threadIdx.x;
  const int wave = tid >> 5;
  const int lane = tid & 31;
  const int l16  = lane & 15;
  const int half = lane >> 4;
  const int row0 = blockIdx.x * 16;

  f32x8 acc[TPW];
#pragma unroll
  for (int t = 0; t < TPW; ++t) acc[t] = f32x8{0.f,0.f,0.f,0.f,0.f,0.f,0.f,0.f};

  const int arow = tid >> 3;                // 0..15
  const int acol = (tid & 7) << 2;          // 0,4,..,28

#pragma unroll
  for (int kt = 0; kt < KT; ++kt) {
    // stage A tile 16x32 f32 -> bf16 in LDS (double buffered, 1 barrier/iter)
    float4 av = *reinterpret_cast<const float4*>(
        A + (size_t)(row0 + arow) * K + (size_t)kt * 32 + acol);
    unsigned long long p =
        (unsigned long long)f2bf(av.x)
      | ((unsigned long long)f2bf(av.y) << 16)
      | ((unsigned long long)f2bf(av.z) << 32)
      | ((unsigned long long)f2bf(av.w) << 48);
    *reinterpret_cast<unsigned long long*>(&As[kt & 1][arow][acol]) = p;
    __syncthreads();

    // A fragment: lane = row l16; K chunks at half*8 and half*8+16 (matches ISA layout)
    Uint4Pair qa;
    qa.a = *reinterpret_cast<const uint4*>(&As[kt & 1][l16][half * 8]);
    qa.b = *reinterpret_cast<const uint4*>(&As[kt & 1][l16][half * 8 + 16]);
    bf16x16 afrag = __builtin_bit_cast(bf16x16, qa);

#pragma unroll
    for (int t = 0; t < TPW; ++t) {
      int nt = wave * TPW + t;
      const bf16x16 bfrag = *reinterpret_cast<const bf16x16*>(
          Bswz + (size_t)(kt * NT + nt) * 512 + lane * 16);
      acc[t] = __builtin_amdgcn_wmma_f32_16x16x32_bf16(
          false, afrag, false, bfrag, (short)0, acc[t], false, false);
    }
  }

  // C layout: VGPR r -> M = half*8 + r, N = l16 (+tile offset)
#pragma unroll
  for (int t = 0; t < TPW; ++t) {
    int n  = (wave * TPW + t) * 16 + l16;
    int mb = half * 8;
#pragma unroll
    for (int r = 0; r < 8; ++r)
      C[(size_t)(row0 + mb + r) * NCOLS + n] = acc[t][r];
  }
}

// ---------------- aggregation ----------------
__global__ void self_init(const float* __restrict__ h0, const float* __restrict__ dinv,
                          float* __restrict__ out, size_t total, int shift) {
  size_t i = (size_t)blockIdx.x * blockDim.x + threadIdx.x;
  if (i < total) {
    float d = dinv[i >> shift];             // self-loop norm = dinv^2
    out[i] = d * d * h0[i];
  }
}

__global__ __launch_bounds__(256) void edge_scatter(
    const float* __restrict__ h0, const long long* __restrict__ eidx,
    const float* __restrict__ dinv, float* __restrict__ out, int E, int F) {
  size_t gid = (size_t)blockIdx.x * blockDim.x + threadIdx.x;
  int per = F >> 2;
  size_t total = (size_t)E * per;
  if (gid >= total) return;
  int e = (int)(gid / per);
  int f = (int)(gid % per) * 4;
  int s = (int)eidx[e];
  int d = (int)eidx[e + E];
  float nrm = dinv[s] * dinv[d];
  float4 v = *reinterpret_cast<const float4*>(h0 + (size_t)s * F + f);
  float* dp = out + (size_t)d * F + f;
  unsafeAtomicAdd(dp + 0, nrm * v.x);       // L2-resident f32 atomics (out fits in 192MB L2)
  unsafeAtomicAdd(dp + 1, nrm * v.y);
  unsafeAtomicAdd(dp + 2, nrm * v.z);
  unsafeAtomicAdd(dp + 3, nrm * v.w);
}

__global__ void bias_act(float* __restrict__ h, const float* __restrict__ b,
                         size_t total, int mask, int relu) {
  size_t i = (size_t)blockIdx.x * blockDim.x + threadIdx.x;
  if (i < total) {
    float v = h[i] + b[i & mask];
    if (relu) v = fmaxf(v, 0.f);
    h[i] = v;
  }
}

// ---------------- decode: wave per edge, 128-dim dot ----------------
__global__ __launch_bounds__(256) void decode(
    const float* __restrict__ z, const long long* __restrict__ pos,
    const long long* __restrict__ neg, float* __restrict__ out, int E) {
  int gw   = blockIdx.x * (blockDim.x >> 5) + (threadIdx.x >> 5);
  int lane = threadIdx.x & 31;
  if (gw >= 2 * E) return;
  const long long* ei = (gw < E) ? pos : neg;
  int e = (gw < E) ? gw : gw - E;
  int a = (int)ei[e];
  int b = (int)ei[e + E];
  const float* za = z + (size_t)a * HID2;
  const float* zb = z + (size_t)b * HID2;
  float s = 0.f;
#pragma unroll
  for (int j = 0; j < 4; ++j) s += za[lane + 32 * j] * zb[lane + 32 * j];
#pragma unroll
  for (int off = 16; off > 0; off >>= 1) s += __shfl_xor(s, off, 32);
  if (lane == 0) out[gw] = s;
}

// ---------------- driver ----------------
extern "C" void kernel_launch(void* const* d_in, const int* in_sizes, int n_in,
                              void* d_out, int out_size, void* d_ws, size_t ws_size,
                              hipStream_t stream) {
  const float*     x      = (const float*)d_in[0];
  const long long* etrain = (const long long*)d_in[1];
  const long long* epos   = (const long long*)d_in[2];
  const long long* eneg   = (const long long*)d_in[3];
  const float*     W1     = (const float*)d_in[4];
  const float*     b1     = (const float*)d_in[5];
  const float*     W2     = (const float*)d_in[6];
  const float*     b2     = (const float*)d_in[7];

  const int nNodes = in_sizes[0] / N_FEAT;  // 50000 (multiple of 16)
  const int E      = in_sizes[1] / 2;       // 800000
  float* out = (float*)d_out;

  char* ws = (char*)d_ws;
  size_t off = 0;
  auto carve = [&](size_t bytes) -> char* {
    char* p = ws + off;
    off += (bytes + 255) & ~(size_t)255;
    return p;
  };
  float*          dinv = (float*)carve((size_t)nNodes * 4);
  unsigned short* W1b  = (unsigned short*)carve((size_t)(HIDDEN/32)*(HIDDEN/16)*512*2);
  unsigned short* W2b  = (unsigned short*)carve((size_t)(HIDDEN/32)*(HID2/16)*512*2);
  float*          bufA = (float*)carve((size_t)nNodes * HIDDEN * 4); // h0, later z0|zpre
  float*          bufB = (float*)carve((size_t)nNodes * HIDDEN * 4); // out1 -> h
  float* z0   = bufA;
  float* zpre = bufA + (size_t)nNodes * HID2;

  const int tb = 256;
  // normalization
  fill_f32 <<<(nNodes + tb - 1) / tb, tb, 0, stream>>>(dinv, 1.0f, nNodes);
  accum_deg<<<(E + tb - 1) / tb, tb, 0, stream>>>(etrain + E, dinv, E);
  rsqrt_ip <<<(nNodes + tb - 1) / tb, tb, 0, stream>>>(dinv, nNodes);
  // weight pre-swizzle (tiny, one-shot per call)
  swizzle_w<<<(HIDDEN/32)*(HIDDEN/16), 32, 0, stream>>>(W1, W1b, HIDDEN, HIDDEN);
  swizzle_w<<<(HIDDEN/32)*(HID2/16),  32, 0, stream>>>(W2, W2b, HIDDEN, HID2);

  // layer 1
  gemm_bf16_wmma<HIDDEN, 4><<<nNodes / 16, 128, 0, stream>>>(x, W1b, bufA);
  size_t tot1 = (size_t)nNodes * HIDDEN;
  self_init   <<<(unsigned)((tot1 + tb - 1) / tb), tb, 0, stream>>>(bufA, dinv, bufB, tot1, 8);
  edge_scatter<<<(unsigned)(((size_t)E*(HIDDEN/4) + tb - 1) / tb), tb, 0, stream>>>(bufA, etrain, dinv, bufB, E, HIDDEN);
  bias_act    <<<(unsigned)((tot1 + tb - 1) / tb), tb, 0, stream>>>(bufB, b1, tot1, HIDDEN - 1, 1);

  // layer 2
  gemm_bf16_wmma<HID2, 2><<<nNodes / 16, 128, 0, stream>>>(bufB, W2b, z0);
  size_t tot2 = (size_t)nNodes * HID2;
  self_init   <<<(unsigned)((tot2 + tb - 1) / tb), tb, 0, stream>>>(z0, dinv, zpre, tot2, 7);
  edge_scatter<<<(unsigned)(((size_t)E*(HID2/4) + tb - 1) / tb), tb, 0, stream>>>(z0, etrain, dinv, zpre, E, HID2);
  bias_act    <<<(unsigned)((tot2 + tb - 1) / tb), tb, 0, stream>>>(zpre, b2, tot2, HID2 - 1, 0);

  // decode: one wave per edge, 2E edges total
  long long nwaves = 2LL * E;
  decode<<<(unsigned)((nwaves * 32 + tb - 1) / tb), tb, 0, stream>>>(zpre, epos, eneg, out, E);
}
